// MHA_18184891531439
// MI455X (gfx1250) — compile-verified, matrix operand order checked by reading, not run on hardware
//
#include <hip/hip_runtime.h>
#include <hip/hip_bf16.h>
#include <stdint.h>

// ---------------------------------------------------------------------------
// Problem constants (from reference): B=2, T=4096, D=768, H=12, hd=64
// ---------------------------------------------------------------------------
#define D_MODEL   768
#define N_HEADS   12
#define HEAD_DIM  64
#define BATCH     2
#define SEQ       4096
#define BT        (BATCH * SEQ)      // 8192 rows
#define BH        (BATCH * N_HEADS)  // 24 head-batches
#define QKV_N     (3 * D_MODEL)      // 2304

typedef __bf16 bf16_t;
typedef __attribute__((ext_vector_type(16))) __bf16 v16bf;
typedef __attribute__((ext_vector_type(8)))  float  v8f;
typedef __attribute__((ext_vector_type(4)))  int    v4i;

#define AS_GLOBAL __attribute__((address_space(1)))
#define AS_LOCAL  __attribute__((address_space(3)))

// gfx1250 async global->LDS DMA (ASYNCcnt-tracked), if this toolchain has it.
// Probe round confirmed: 4 args, param0 is v4i in the global address space.
#if defined(__has_builtin)
#if __has_builtin(__builtin_amdgcn_global_load_async_to_lds_b128)
#define HAVE_ASYNC_LDS 1
#endif
#endif
#ifndef HAVE_ASYNC_LDS
#define HAVE_ASYNC_LDS 0
#endif

// ---------------------------------------------------------------------------
// Helpers
// ---------------------------------------------------------------------------
__device__ __forceinline__ unsigned short f2bf(float f) {
    union { float f; unsigned u; } c; c.f = f;
    unsigned u = c.u;
    u = u + 0x7FFFu + ((u >> 16) & 1u);   // round-to-nearest-even
    return (unsigned short)(u >> 16);
}

__device__ __forceinline__ v8f vzero8() {
    v8f z;
#pragma unroll
    for (int i = 0; i < 8; ++i) z[i] = 0.0f;
    return z;
}

union FragBits { uint4 q[2]; v16bf v; };

// Load one 16x32 bf16 WMMA A/B fragment from a row-major (k-contiguous)
// matrix. Per ISA 7.12.2 (16-bit A-matrix 16x32): lanes 0-15 hold
// K = {0..7, 16..23}, lanes 16-31 hold K = {8..15, 24..31}; row = lane%16.
// Both chunks are contiguous 16-byte loads.
__device__ __forceinline__ v16bf load_frag16x32(const unsigned short* base,
                                                int ld, int lane) {
    const int half = (lane >> 4) & 1;
    const int m    = lane & 15;
    const unsigned short* r = base + (size_t)m * ld + half * 8;
    FragBits f;
    f.q[0] = *(const uint4*)(r);        // K = half*8 .. half*8+7
    f.q[1] = *(const uint4*)(r + 16);   // K = 16 + half*8 .. +7
    return f.v;
}

__device__ __forceinline__ v8f wmma_bf16(v16bf a, v16bf b, v8f c) {
    return __builtin_amdgcn_wmma_f32_16x16x32_bf16(
        /*neg_a=*/false, a, /*neg_b=*/false, b,
        /*c_mod=*/(short)0, c, /*reuse_a=*/false, /*reuse_b=*/false);
}

// ---------------------------------------------------------------------------
// Kernel 1a: f32 -> bf16 conversion (4 elements / thread)
// ---------------------------------------------------------------------------
__global__ void cvt_f32_to_bf16(const float* __restrict__ src,
                                unsigned short* __restrict__ dst, int n4) {
    int i = blockIdx.x * blockDim.x + threadIdx.x;
    if (i >= n4) return;
    float4 f = ((const float4*)src)[i];
    unsigned v0 = (unsigned)f2bf(f.x) | ((unsigned)f2bf(f.y) << 16);
    unsigned v1 = (unsigned)f2bf(f.z) | ((unsigned)f2bf(f.w) << 16);
    ((uint2*)dst)[i] = make_uint2(v0, v1);
}

// ---------------------------------------------------------------------------
// Kernel 1b: f32 [K][N] -> bf16 transposed [N][K]
// (weights pre-transposed so GEMM B-tile staging is 128-bit contiguous)
// ---------------------------------------------------------------------------
__global__ void cvt_transpose_bf16(const float* __restrict__ src,
                                   unsigned short* __restrict__ dst,
                                   int K, int N) {
    int i = blockIdx.x * blockDim.x + threadIdx.x;
    if (i >= K * N) return;
    int k = i / N, n = i % N;
    dst[(size_t)n * K + k] = f2bf(src[i]);
}

// ---------------------------------------------------------------------------
// Kernel 2/4: tiled bf16 WMMA GEMM  C[M,N] = A[M,K] * Bt[N,K]^T + bias
//   block tile 128x128, 256 threads = 8 waves (2 along M x 4 along N),
//   each wave computes 64x32 via 4x2 WMMA accumulators.
//   LDS double-buffered; tile k+1 staged via async global->LDS DMA while
//   the WMMAs consume tile k; published with s_wait_asynccnt + barrier.
// MODE 0: QKV epilogue -> scatter Q (scaled 1/8), K [bh,t,64], V^T [bh,64,t]
// MODE 1: bias + f32 store to out
// ---------------------------------------------------------------------------
template <int MODE>
__global__ __launch_bounds__(256)
void gemm_bf16_wmma(const unsigned short* __restrict__ A,   // [M][K]
                    const unsigned short* __restrict__ Bt,  // [N][K]
                    const float* __restrict__ bias,
                    float* __restrict__ outF,
                    unsigned short* __restrict__ Qb,
                    unsigned short* __restrict__ Kb,
                    unsigned short* __restrict__ Vt,
                    int M, int N, int K) {
    __shared__ __align__(16) unsigned short lA[2][128 * 32];  // [row][k]
    __shared__ __align__(16) unsigned short lB[2][128 * 32];  // [col][k]

    const int tid  = threadIdx.x;
    const int lane = tid & 31;
    const int wave = tid >> 5;
    const int wm   = (wave & 1) * 64;   // wave row offset within tile
    const int wn   = (wave >> 1) * 32;  // wave col offset within tile
    const int half = (lane >> 4) & 1;
    const int l16  = lane & 15;

    const int rowBase = blockIdx.x * 128;
    const int colBase = blockIdx.y * 128;

    auto stage = [&](int buf, int k0) {
#pragma unroll
        for (int i = 0; i < 2; ++i) {
            int chunk = tid + i * 256;          // 0..511
            int row   = chunk >> 2;
            int c8    = (chunk & 3) * 8;
#if HAVE_ASYNC_LDS
            __builtin_amdgcn_global_load_async_to_lds_b128(
                (AS_GLOBAL v4i*)&A[(size_t)(rowBase + row) * K + k0 + c8],
                (AS_LOCAL  v4i*)&lA[buf][row * 32 + c8], 0, 0);
            __builtin_amdgcn_global_load_async_to_lds_b128(
                (AS_GLOBAL v4i*)&Bt[(size_t)(colBase + row) * K + k0 + c8],
                (AS_LOCAL  v4i*)&lB[buf][row * 32 + c8], 0, 0);
#else
            *(uint4*)&lA[buf][row * 32 + c8] =
                *(const uint4*)&A[(size_t)(rowBase + row) * K + k0 + c8];
            *(uint4*)&lB[buf][row * 32 + c8] =
                *(const uint4*)&Bt[(size_t)(colBase + row) * K + k0 + c8];
#endif
        }
    };

    v8f acc[4][2];
#pragma unroll
    for (int mi = 0; mi < 4; ++mi)
#pragma unroll
        for (int ni = 0; ni < 2; ++ni) acc[mi][ni] = vzero8();

    stage(0, 0);
#if HAVE_ASYNC_LDS
    asm volatile("s_wait_asynccnt 0" ::: "memory");
#endif
    __syncthreads();

    int cur = 0;
    for (int k0 = 0; k0 < K; k0 += 32) {
        if (k0 + 32 < K) stage(cur ^ 1, k0 + 32);  // overlap with WMMAs below

        v16bf aF[4], bF[2];
#pragma unroll
        for (int mi = 0; mi < 4; ++mi)
            aF[mi] = load_frag16x32(&lA[cur][(wm + mi * 16) * 32], 32, lane);
#pragma unroll
        for (int ni = 0; ni < 2; ++ni)
            bF[ni] = load_frag16x32(&lB[cur][(wn + ni * 16) * 32], 32, lane);

#pragma unroll
        for (int mi = 0; mi < 4; ++mi)
#pragma unroll
            for (int ni = 0; ni < 2; ++ni)
                acc[mi][ni] = wmma_bf16(aF[mi], bF[ni], acc[mi][ni]);

#if HAVE_ASYNC_LDS
        asm volatile("s_wait_asynccnt 0" ::: "memory");
#endif
        __syncthreads();
        cur ^= 1;
    }

    // --- epilogue; C layout: VGPR r holds row M = r + 8*half, col N = lane%16
#pragma unroll
    for (int mi = 0; mi < 4; ++mi) {
#pragma unroll
        for (int ni = 0; ni < 2; ++ni) {
#pragma unroll
            for (int r = 0; r < 8; ++r) {
                int row = rowBase + wm + mi * 16 + r + 8 * half;
                int col = colBase + wn + ni * 16 + l16;
                float val = acc[mi][ni][r] + bias[col];
                if (MODE == 0) {
                    int part = col / D_MODEL;       // 0=Q 1=K 2=V
                    int cc   = col % D_MODEL;
                    int h    = cc >> 6;
                    int d    = cc & 63;
                    int b    = row >> 12;           // /4096
                    int t    = row & 4095;
                    int bh   = b * N_HEADS + h;
                    if (part == 0) {
                        // fold the 1/sqrt(hd) score scale into Q
                        Qb[((size_t)bh * SEQ + t) * HEAD_DIM + d] =
                            f2bf(val * 0.125f);
                    } else if (part == 1) {
                        Kb[((size_t)bh * SEQ + t) * HEAD_DIM + d] = f2bf(val);
                    } else {
                        // V transposed: [bh, d, t] so attn B-frags are contiguous
                        Vt[((size_t)bh * HEAD_DIM + d) * SEQ + t] = f2bf(val);
                    }
                } else {
                    outF[(size_t)row * N + col] = val;
                }
            }
        }
    }
}

// ---------------------------------------------------------------------------
// Kernel 3: causal flash attention.
//   grid = (T/128, B*H); block = 128 (4 waves). Wave w owns 32 query rows
//   [qTile*128 + w*32, +32) as 2 M-frags, so each K/V fragment feeds 2 WMMAs.
//   Online softmax; l kept as per-lane partial, reduced once at the end.
// ---------------------------------------------------------------------------
__global__ __launch_bounds__(128)
void flash_attn_wmma(const unsigned short* __restrict__ Qb,
                     const unsigned short* __restrict__ Kb,
                     const unsigned short* __restrict__ Vt,
                     unsigned short* __restrict__ Ob) {
    __shared__ __align__(16) unsigned short lP[4][32 * 64];  // wave-private P

    const int tid  = threadIdx.x;
    const int lane = tid & 31;
    const int wave = tid >> 5;
    const int half = (lane >> 4) & 1;
    const int l16  = lane & 15;

    const int bh    = blockIdx.y;
    const int qTile = blockIdx.x;
    const int qRow0 = qTile * 128 + wave * 32;

    // Q fragments (Q pre-scaled by 1/sqrt(hd))
    const unsigned short* Qbase = Qb + ((size_t)bh * SEQ + qRow0) * HEAD_DIM;
    v16bf qA[2][2];
#pragma unroll
    for (int mi = 0; mi < 2; ++mi) {
        qA[mi][0] = load_frag16x32(Qbase + mi * 16 * HEAD_DIM,      HEAD_DIM, lane);
        qA[mi][1] = load_frag16x32(Qbase + mi * 16 * HEAD_DIM + 32, HEAD_DIM, lane);
    }

    float mrow[2][8], lrow[2][8];   // lrow: per-lane partial row sums
    v8f o[2][4];
#pragma unroll
    for (int mi = 0; mi < 2; ++mi) {
#pragma unroll
        for (int r = 0; r < 8; ++r) { mrow[mi][r] = -1.0e30f; lrow[mi][r] = 0.0f; }
#pragma unroll
        for (int f = 0; f < 4; ++f) o[mi][f] = vzero8();
    }

    const int nkt = 2 * (qTile + 1);  // 64-wide key tiles covering causal range
    for (int kt = 0; kt < nkt; ++kt) {
        const int k0t = kt * 64;
        if (k0t > qRow0 + 31) break;  // fully above diagonal for this wave

        // prefetch next key/value tiles into cache
        if (kt + 1 < nkt) {
            __builtin_prefetch(Kb + ((size_t)bh * SEQ + k0t + 64) * HEAD_DIM +
                                   (size_t)lane * 128, 0, 1);
            __builtin_prefetch(Vt + ((size_t)bh * HEAD_DIM + (lane & 63)) * SEQ +
                                   k0t + 64, 0, 1);
        }

        // ---- S = Q * K^T (32x64 per wave) ----
        v8f s[2][4];
#pragma unroll
        for (int mi = 0; mi < 2; ++mi)
#pragma unroll
            for (int f = 0; f < 4; ++f) s[mi][f] = vzero8();
#pragma unroll
        for (int f = 0; f < 4; ++f) {
            const unsigned short* Kbase =
                Kb + ((size_t)bh * SEQ + k0t + f * 16) * HEAD_DIM;
            v16bf kB0 = load_frag16x32(Kbase,      HEAD_DIM, lane);
            v16bf kB1 = load_frag16x32(Kbase + 32, HEAD_DIM, lane);
#pragma unroll
            for (int mi = 0; mi < 2; ++mi) {
                s[mi][f] = wmma_bf16(qA[mi][0], kB0, s[mi][f]);
                s[mi][f] = wmma_bf16(qA[mi][1], kB1, s[mi][f]);
            }
        }

        // ---- causal mask + online softmax per M-frag ----
#pragma unroll
        for (int mi = 0; mi < 2; ++mi) {
            const int qmin = qRow0 + mi * 16;
            if (k0t + 63 > qmin) {  // tile touches the diagonal for this frag
#pragma unroll
                for (int f = 0; f < 4; ++f)
#pragma unroll
                    for (int r = 0; r < 8; ++r) {
                        int qg = qmin + r + 8 * half;
                        int kg = k0t + f * 16 + l16;
                        if (kg > qg) s[mi][f][r] = -1.0e30f;
                    }
            }
#pragma unroll
            for (int r = 0; r < 8; ++r) {
                float v = fmaxf(fmaxf(s[mi][0][r], s[mi][1][r]),
                                fmaxf(s[mi][2][r], s[mi][3][r]));
#pragma unroll
                for (int d = 1; d < 16; d <<= 1)
                    v = fmaxf(v, __shfl_xor(v, d, 32));
                float mnew  = fmaxf(mrow[mi][r], v);
                float alpha = __expf(mrow[mi][r] - mnew);
                mrow[mi][r] = mnew;
                lrow[mi][r] *= alpha;
#pragma unroll
                for (int f = 0; f < 4; ++f) o[mi][f][r] *= alpha;
#pragma unroll
                for (int f = 0; f < 4; ++f) {
                    float p = __expf(s[mi][f][r] - mnew);
                    s[mi][f][r] = p;
                    lrow[mi][r] += p;   // per-lane partial; reduced at the end
                }
            }
        }

        // ---- P: C-layout regs -> LDS -> A-layout frags (wave-private) ----
#pragma unroll
        for (int mi = 0; mi < 2; ++mi)
#pragma unroll
            for (int f = 0; f < 4; ++f)
#pragma unroll
                for (int r = 0; r < 8; ++r)
                    lP[wave][(mi * 16 + r + 8 * half) * 64 + f * 16 + l16] =
                        f2bf(s[mi][f][r]);

        asm volatile("s_wait_dscnt 0" ::: "memory");

        v16bf pA[2][2];
#pragma unroll
        for (int mi = 0; mi < 2; ++mi) {
            pA[mi][0] = load_frag16x32(&lP[wave][mi * 16 * 64],      64, lane);
            pA[mi][1] = load_frag16x32(&lP[wave][mi * 16 * 64 + 32], 64, lane);
        }

        // ---- O += P * V  (V stored transposed: [bh, d, t]) ----
#pragma unroll
        for (int f = 0; f < 4; ++f) {
            const unsigned short* Vbase =
                Vt + ((size_t)bh * HEAD_DIM + f * 16) * SEQ + k0t;
            v16bf vB0 = load_frag16x32(Vbase,      SEQ, lane);  // keys 0..31
            v16bf vB1 = load_frag16x32(Vbase + 32, SEQ, lane);  // keys 32..63
#pragma unroll
            for (int mi = 0; mi < 2; ++mi) {
                o[mi][f] = wmma_bf16(pA[mi][0], vB0, o[mi][f]);
                o[mi][f] = wmma_bf16(pA[mi][1], vB1, o[mi][f]);
            }
        }
    }

    // ---- final cross-lane row-sum reduction, normalize, store bf16 ----
    const int b = bh / N_HEADS;
    const int h = bh % N_HEADS;
#pragma unroll
    for (int mi = 0; mi < 2; ++mi)
#pragma unroll
        for (int r = 0; r < 8; ++r) {
            float lsum = lrow[mi][r];
#pragma unroll
            for (int d = 1; d < 16; d <<= 1) lsum += __shfl_xor(lsum, d, 32);
            float inv = 1.0f / lsum;
            int t = qRow0 + mi * 16 + r + 8 * half;
#pragma unroll
            for (int f = 0; f < 4; ++f)
                Ob[((size_t)(b * SEQ + t)) * D_MODEL + h * HEAD_DIM + f * 16 + l16] =
                    f2bf(o[mi][f][r] * inv);
        }
}

// ---------------------------------------------------------------------------
// Launch: cvt -> QKV GEMM -> flash attention -> output GEMM
// ---------------------------------------------------------------------------
extern "C" void kernel_launch(void* const* d_in, const int* in_sizes, int n_in,
                              void* d_out, int out_size, void* d_ws,
                              size_t ws_size, hipStream_t stream) {
    const float* x     = (const float*)d_in[0];
    const float* W_qkv = (const float*)d_in[1];
    const float* b_qkv = (const float*)d_in[2];
    const float* W_o   = (const float*)d_in[3];
    const float* b_o   = (const float*)d_in[4];
    float* out = (float*)d_out;

    unsigned short* ws = (unsigned short*)d_ws;
    size_t off = 0;
    unsigned short* xbf  = ws + off; off += (size_t)BT * D_MODEL;        // 6.29M
    unsigned short* Wqbf = ws + off; off += (size_t)D_MODEL * QKV_N;     // [N][K]
    unsigned short* Wobf = ws + off; off += (size_t)D_MODEL * D_MODEL;   // [N][K]
    unsigned short* Qbf  = ws + off; off += (size_t)BH * SEQ * HEAD_DIM;
    unsigned short* Kbf  = ws + off; off += (size_t)BH * SEQ * HEAD_DIM;
    unsigned short* Vt   = ws + off; off += (size_t)BH * HEAD_DIM * SEQ;
    unsigned short* Obf  = xbf;  // reuse xbf (dead after QKV GEMM)

    // 1) conversions (x plain; weights transposed to [N][K])
    {
        int n4 = (BT * D_MODEL) / 4;
        cvt_f32_to_bf16<<<(n4 + 255) / 256, 256, 0, stream>>>(x, xbf, n4);
        int n = D_MODEL * QKV_N;
        cvt_transpose_bf16<<<(n + 255) / 256, 256, 0, stream>>>(W_qkv, Wqbf,
                                                                D_MODEL, QKV_N);
        n = D_MODEL * D_MODEL;
        cvt_transpose_bf16<<<(n + 255) / 256, 256, 0, stream>>>(W_o, Wobf,
                                                                D_MODEL, D_MODEL);
    }

    // 2) QKV projection: [8192,768] x [768,2304] + bias, scatter Q/K/V^T
    gemm_bf16_wmma<0><<<dim3(BT / 128, QKV_N / 128), 256, 0, stream>>>(
        xbf, Wqbf, b_qkv, nullptr, Qbf, Kbf, Vt, BT, QKV_N, D_MODEL);

    // 3) causal flash attention -> Obf [8192, 768] bf16
    flash_attn_wmma<<<dim3(SEQ / 128, BH), 128, 0, stream>>>(Qbf, Kbf, Vt, Obf);

    // 4) output projection: [8192,768] x [768,768] + bias -> f32 out
    gemm_bf16_wmma<1><<<dim3(BT / 128, D_MODEL / 128), 256, 0, stream>>>(
        Obf, Wobf, b_o, out, nullptr, nullptr, nullptr, BT, D_MODEL, D_MODEL);
}
